// Bary_layer_41566693491079
// MI455X (gfx1250) — matrix operand order repaired and checked by reading, compiled, not run on hardware
//
#include <hip/hip_runtime.h>
#include <hip/hip_bf16.h>

typedef __attribute__((ext_vector_type(16))) _Float16 v16h;
typedef __attribute__((ext_vector_type(8)))  _Float16 v8h;
typedef __attribute__((ext_vector_type(8)))  float    v8f;

#define D    64
#define DEG  16
#define WPB  8          // nodes (waves) per block
#define EPSV 1e-30f

__device__ __forceinline__ float fastrcp(float x) { return __builtin_amdgcn_rcpf(x); }
// All producer->consumer LDS traffic in kernel 1 is intra-wave (per-wave slices),
// and same-wave LDS ops complete in order on CDNA5 -> a compiler-only fence suffices.
__device__ __forceinline__ void wavefence() { __builtin_amdgcn_wave_barrier(); }

// ---- B operand (32x16, f16): lane(half,m): e0..7 -> K = kb + 8*half + e, e8..15 -> K = kb + 16 + 8*half + (e-8), N = m.
// b stored [m][k] row-major => two 8-halfword runs 16 apart = two 16B LDS loads.
__device__ __forceinline__ v16h load_bop(const _Float16* bL, int kb, int hf, int m) {
  const _Float16* p = bL + m * D + kb + 8 * hf;
  v8h lo = *(const v8h*)p;
  v8h hi = *(const v8h*)(p + 16);
  return __builtin_shufflevector(lo, hi, 0,1,2,3,4,5,6,7,8,9,10,11,12,13,14,15);
}

// ---- A operand (16x32, f16): rows 16t..16t+15 of K, cols kb..kb+31.
// lanes 0-15 : e0..7 -> cols kb+0..7,  e8..15 -> cols kb+16..23
// lanes16-31 : e0..7 -> cols kb+8..15, e8..15 -> cols kb+24..31
// => per lane two contiguous 8-halfword runs at (kb+8*half) and (kb+16+8*half).
__device__ __forceinline__ v16h load_aop(const _Float16* Kp, int t, int kb, int hf, int m) {
  const _Float16* p = Kp + (size_t)(16 * t + m) * D + kb + 8 * hf;
  v8h lo = *(const v8h*)p;
  v8h hi = *(const v8h*)(p + 16);
  return __builtin_shufflevector(lo, hi, 0,1,2,3,4,5,6,7,8,9,10,11,12,13,14,15);
}

// ---------------- kernel 0: Gibbs kernel K = exp(-C/REG) and K^T in f16 ----------------
__global__ void prep_kernel(const float* __restrict__ C,
                            _Float16* __restrict__ Kf, _Float16* __restrict__ Ktf) {
  const int idx = blockIdx.x * blockDim.x + threadIdx.x;
  if (idx < D * D) {
    const int i = idx >> 6, j = idx & 63;
    const float k = expf(-10.0f * C[idx]);   // 1/REG = 10
    Kf[idx] = (_Float16)k;
    Ktf[j * D + i] = (_Float16)k;
  }
}

// ---------------- kernel 1: per-node Sinkhorn barycenter (WMMA) ----------------
__global__ __launch_bounds__(256) void bary_sinkhorn_kernel(
    const float* __restrict__ U, const int* __restrict__ adj,
    const _Float16* __restrict__ Kf, const _Float16* __restrict__ Ktf,
    float* __restrict__ out, int n)
{
  __shared__ __align__(16) _Float16 sh_a[WPB][16 * D];   // A[m][d]  (f16)
  __shared__ __align__(16) _Float16 sh_b[WPB][16 * D];   // b/ratio [m][d] (f16)
  __shared__ float sh_bary[WPB][D];

  const int lane = threadIdx.x & 31;
  const int wave = threadIdx.x >> 5;
  const int hf = lane >> 4;       // half-wave index
  const int m  = lane & 15;       // neighbor / N index
  int node = blockIdx.x * WPB + wave;
  const bool writer = node < n;
  if (!writer) node = n - 1;      // clamp: keep EXEC all-ones through WMMAs

  _Float16* aL = sh_a[wave];
  _Float16* bL = sh_b[wave];
  float*  baryL = sh_bary[wave];

  // ---- preload K / K^T A-operand tiles into registers (constant across iters) ----
  v16h Ka[4][2], Kta[4][2];
#pragma unroll
  for (int t = 0; t < 4; ++t)
#pragma unroll
    for (int c = 0; c < 2; ++c) {
      Ka[t][c]  = load_aop(Kf,  t, 32 * c, hf, m);
      Kta[t][c] = load_aop(Ktf, t, 32 * c, hf, m);
    }

  // ---- A(:,m) = normalize(exp(U[adj[node,m],:]));  also init b = 1 ----
  {
    const int j = adj[(size_t)node * DEG + m];
    const float* urow = U + (size_t)j * D + hf * 32;
    float vals[32];
    float s = 0.f;
#pragma unroll
    for (int i = 0; i < 32; ++i) { vals[i] = __expf(urow[i]); s += vals[i]; }
    s += __shfl_xor(s, 16);       // combine the two half-rows
    const float inv = fastrcp(s);
#pragma unroll
    for (int c = 0; c < 4; ++c) {
      v8h pk, one;
#pragma unroll
      for (int v = 0; v < 8; ++v) {
        pk[v]  = (_Float16)(vals[c * 8 + v] * inv);
        one[v] = (_Float16)1.0f;
      }
      *(v8h*)(aL + m * D + hf * 32 + c * 8) = pk;
      *(v8h*)(bL + m * D + hf * 32 + c * 8) = one;
    }
  }
  wavefence();

  v8f acc[4];
  for (int it = 0; it < 5; ++it) {
    const bool last = (it == 4);
    // ---- Kb = K @ b ----
    v16h B0 = load_bop(bL, 0, hf, m);
    v16h B1 = load_bop(bL, 32, hf, m);
#pragma unroll
    for (int t = 0; t < 4; ++t) {
      v8f z = {};
      z = __builtin_amdgcn_wmma_f32_16x16x32_f16(false, Ka[t][0], false, B0, (short)0, z, false, false);
      z = __builtin_amdgcn_wmma_f32_16x16x32_f16(false, Ka[t][1], false, B1, (short)0, z, false, false);
      acc[t] = z;
    }
    wavefence();
    // ---- ratio = A * rcp(Kb + eps) -> bL ----  (D layout: d = 16t + v + 8*hf, col m)
#pragma unroll
    for (int t = 0; t < 4; ++t) {
      const v8h av = *(const v8h*)(aL + m * D + 16 * t + 8 * hf);
      v8h r;
#pragma unroll
      for (int v = 0; v < 8; ++v)
        r[v] = (_Float16)((float)av[v] * fastrcp(acc[t][v] + EPSV));
      *(v8h*)(bL + m * D + 16 * t + 8 * hf) = r;
    }
    wavefence();
    // ---- phi = K^T @ ratio ----
    B0 = load_bop(bL, 0, hf, m);
    B1 = load_bop(bL, 32, hf, m);
#pragma unroll
    for (int t = 0; t < 4; ++t) {
      v8f z = {};
      z = __builtin_amdgcn_wmma_f32_16x16x32_f16(false, Kta[t][0], false, B0, (short)0, z, false, false);
      z = __builtin_amdgcn_wmma_f32_16x16x32_f16(false, Kta[t][1], false, B1, (short)0, z, false, false);
      acc[t] = z;
    }
    // ---- bary = exp(mean_m log(phi+eps));  b = bary * rcp(phi+eps) ----
#pragma unroll
    for (int t = 0; t < 4; ++t) {
      float ph[8], lp[8];
#pragma unroll
      for (int v = 0; v < 8; ++v) { ph[v] = acc[t][v] + EPSV; lp[v] = __logf(ph[v]); }
#pragma unroll
      for (int mask = 1; mask <= 8; mask <<= 1)
#pragma unroll
        for (int v = 0; v < 8; ++v) lp[v] += __shfl_xor(lp[v], mask);  // sum over 16 m-lanes
      v8h nb;
#pragma unroll
      for (int v = 0; v < 8; ++v) {
        const float bar = __expf(lp[v] * 0.0625f);     // geometric mean, w = 1/16
        nb[v] = (_Float16)(bar * fastrcp(ph[v]));
        if (last && m == 0) baryL[16 * t + 8 * hf + v] = bar;
      }
      *(v8h*)(bL + m * D + 16 * t + 8 * hf) = nb;
    }
    wavefence();
  }

  if (writer) {
    out[(size_t)node * D + lane]      = __logf(baryL[lane] + EPSV);
    out[(size_t)node * D + lane + 32] = __logf(baryL[lane + 32] + EPSV);
  }
}

// ---------------- kernel 2: partial Gram matrices G = out^T out ----------------
__global__ __launch_bounds__(256) void gram_kernel(const float* __restrict__ out,
                                                   float* __restrict__ partials, int n) {
  __shared__ float rowb[4][D];
  const int tid = threadIdx.x;
  const int i0 = tid >> 2;            // G row
  const int j0 = (tid & 3) << 4;      // G col group
  const int lr = tid >> 6;            // staging row (0..3)
  const int lc = tid & 63;            // staging col
  float acc[16];
#pragma unroll
  for (int q = 0; q < 16; ++q) acc[q] = 0.f;
  for (int r0 = blockIdx.x * 4; r0 < n; r0 += gridDim.x * 4) {
    __syncthreads();
    const int rr = r0 + lr;
    rowb[lr][lc] = (rr < n) ? out[(size_t)rr * D + lc] : 0.f;
    __syncthreads();
#pragma unroll
    for (int r = 0; r < 4; ++r) {
      const float xi = rowb[r][i0];
#pragma unroll
      for (int q = 0; q < 16; ++q) acc[q] = fmaf(xi, rowb[r][j0 + q], acc[q]);
    }
  }
  float* P = partials + (size_t)blockIdx.x * (D * D);
#pragma unroll
  for (int q = 0; q < 16; ++q) P[i0 * D + j0 + q] = acc[q];
}

// ---------------- kernel 3: reduce partials, Cholesky (G = R^T R), R^-1 ----------------
__global__ __launch_bounds__(256) void chol_kernel(const float* __restrict__ partials,
                                                   int nb, float* __restrict__ Rinv) {
  __shared__ float G[D * D];
  __shared__ float X[D * D];
  const int tid = threadIdx.x;
  for (int idx = tid; idx < D * D; idx += 256) {
    float s = 0.f;
    for (int b = 0; b < nb; ++b) s += partials[(size_t)b * (D * D) + idx];
    G[idx] = s;
    X[idx] = 0.f;
  }
  __syncthreads();
  // Right-looking Cholesky; upper factor R in upper triangle of G, diag(R) > 0.
  for (int k = 0; k < D; ++k) {
    if (tid == 0) G[k * D + k] = sqrtf(G[k * D + k]);
    __syncthreads();
    const float dk = G[k * D + k];
    if (tid > k && tid < D) G[k * D + tid] /= dk;
    __syncthreads();
    for (int idx = tid; idx < D * D; idx += 256) {
      const int i = idx >> 6, j = idx & 63;
      if (i > k && j >= i) G[idx] -= G[k * D + i] * G[k * D + j];
    }
    __syncthreads();
  }
  // Column-parallel back-substitution: X = R^-1 (upper triangular)
  if (tid < D) {
    const int j = tid;
    X[j * D + j] = 1.f / G[j * D + j];
    for (int i = j - 1; i >= 0; --i) {
      float s = 0.f;
      for (int t = i + 1; t <= j; ++t) s += G[i * D + t] * X[t * D + j];
      X[i * D + j] = -s / G[i * D + i];
    }
  }
  __syncthreads();
  for (int idx = tid; idx < D * D; idx += 256) Rinv[idx] = X[idx];
}

// ---------------- kernel 4: Q = out @ R^-1, in-place on d_out ----------------
__global__ __launch_bounds__(256) void apply_kernel(float* __restrict__ out,
                                                    const float* __restrict__ Rinv, int n) {
  __shared__ float Rs[D * D];
  const int tid = threadIdx.x;
  for (int idx = tid; idx < D * D; idx += 256) Rs[idx] = Rinv[idx];
  __syncthreads();
  const int lane = tid & 31;
  const int wave = tid >> 5;
  const int row = blockIdx.x * WPB + wave;
  if (row >= n) return;
  float* orow = out + (size_t)row * D;
  const float e0 = orow[lane];
  const float e1 = orow[lane + 32];
  float q0 = 0.f, q1 = 0.f;
  for (int k = 0; k < D; ++k) {
    const float xk = (k < 32) ? __shfl(e0, k) : __shfl(e1, k - 32);
    q0 = fmaf(xk, Rs[k * D + lane], q0);
    q1 = fmaf(xk, Rs[k * D + lane + 32], q1);
  }
  orow[lane] = q0;
  orow[lane + 32] = q1;
}

extern "C" void kernel_launch(void* const* d_in, const int* in_sizes, int n_in,
                              void* d_out, int out_size, void* d_ws, size_t ws_size,
                              hipStream_t stream) {
  const float* U   = (const float*)d_in[0];   // [n, 64]
  const float* C   = (const float*)d_in[1];   // [64, 64]
  const int*   adj = (const int*)d_in[2];     // [n, 16]
  float* out = (float*)d_out;                 // [n, 64]
  const int n = in_sizes[0] / D;

  // workspace layout
  char* ws = (char*)d_ws;
  _Float16* Kf       = (_Float16*)(ws);             //  8 KB
  _Float16* Ktf      = (_Float16*)(ws + 8192);      //  8 KB
  float*    Rinv     = (float*)(ws + 16384);        // 16 KB
  float*    partials = (float*)(ws + 32768);        // nb * 16 KB

  int nb = 1;
  if (ws_size > (size_t)(32768 + D * D * 4)) {
    size_t cap = (ws_size - 32768) / (size_t)(D * D * 4);
    nb = (int)(cap < 64 ? cap : 64);
    if (nb < 1) nb = 1;
  }

  const int nblk = (n + WPB - 1) / WPB;
  hipLaunchKernelGGL(prep_kernel, dim3(16), dim3(256), 0, stream, C, Kf, Ktf);
  hipLaunchKernelGGL(bary_sinkhorn_kernel, dim3(nblk), dim3(256), 0, stream,
                     U, adj, Kf, Ktf, out, n);
  hipLaunchKernelGGL(gram_kernel, dim3(nb), dim3(256), 0, stream, out, partials, n);
  hipLaunchKernelGGL(chol_kernel, dim3(1), dim3(256), 0, stream, partials, nb, Rinv);
  hipLaunchKernelGGL(apply_kernel, dim3(nblk), dim3(256), 0, stream, out, Rinv, n);
  (void)n_in; (void)out_size;
}
